// SupConLoss_37538014167620
// MI455X (gfx1250) — matrix operand order repaired and checked by reading, not compiled, not run on hardware
//
#include <hip/hip_runtime.h>
#include <hip/hip_bf16.h>

typedef __attribute__((ext_vector_type(16))) _Float16 v16h;
typedef __attribute__((ext_vector_type(8)))  _Float16 v8h;
typedef __attribute__((ext_vector_type(4)))  _Float16 v4h;
typedef __attribute__((ext_vector_type(8)))  float    v8f;
typedef __attribute__((ext_vector_type(4)))  float    v4f;

#define NROWS 8192   // N = B * n_views
#define BATCH 4096   // B (power of two -> mask with & (BATCH-1))
#define DIM   128
#define CTILE 64     // column tile held in LDS
#define INV_T 14.285714285714286f            // 1 / 0.07
#define K2    20.60992915555662f             // (1/0.07) * log2(e)

// ---------------------------------------------------------------------------
// Kernel 1: L2-normalize each row of feat [N, D] (f32) and store as f16.
// One wave per row: 32 lanes x float4 = 128 elements.
// ---------------------------------------------------------------------------
__global__ __launch_bounds__(256) void supcon_normalize(
    const float* __restrict__ in, _Float16* __restrict__ out) {
  const int wave = threadIdx.x >> 5;
  const int lane = threadIdx.x & 31;
  const int row  = blockIdx.x * 8 + wave;
  const float* rp = in + (size_t)row * DIM + lane * 4;
  v4f v = *(const v4f*)rp;
  float ss = v[0]*v[0] + v[1]*v[1] + v[2]*v[2] + v[3]*v[3];
  #pragma unroll
  for (int m = 16; m >= 1; m >>= 1) ss += __shfl_xor(ss, m, 32);
  const float scale = 1.0f / fmaxf(sqrtf(ss), 1e-12f);
  v4h o;
  o[0] = (_Float16)(v[0]*scale); o[1] = (_Float16)(v[1]*scale);
  o[2] = (_Float16)(v[2]*scale); o[3] = (_Float16)(v[3]*scale);
  *(v4h*)(out + (size_t)row * DIM + lane * 4) = o;
}

// ---------------------------------------------------------------------------
// Kernel 2: fused Gram-matrix + masked online-softmax pass.
// grid = 64 blocks x 256 threads (8 waves). Each wave owns a 16-row A block
// (registers, 4x v16h) and sweeps all 8192 columns in 64-col LDS tiles with
// v_wmma_f32_16x16x32_f16. Per-row accumulators (shift = 1/T = exact row max
// of a normalized Gram matrix):
//   denom  = sum_{j!=i} exp(s_ij - 1/T)
//   possum = sum_{j!=i, label match} s_ij
//   cnt    = #{j!=i, label match}
// Epilogue is fully branchless (v_cndmask + v_fma) so EXEC stays all-ones
// and the exp/FMA stream can co-execute with the WMMA (XDL) pipeline.
// ---------------------------------------------------------------------------
__global__ __launch_bounds__(256) void supcon_main(
    const _Float16* __restrict__ f16,
    const int* __restrict__ labels,
    float* __restrict__ rowLoss) {
  __shared__ __align__(32) _Float16 Bsh[CTILE * DIM];
  __shared__ int labSh[CTILE];

  const int t    = threadIdx.x;
  const int wave = t >> 5;
  const int lane = t & 31;
  const int l2   = lane & 15;
  const int hi   = lane >> 4;
  const int rowBase = blockIdx.x * 128 + wave * 16;

  // ---- A block: 16 rows x 128 K, contiguous k order, same as B ----
  // lane l (0-15): row M=l, halves 0..15 = K[16*hi .. +15] (contiguous)
  v16h a[4];
  {
    const _Float16* rp = f16 + (size_t)(rowBase + l2) * DIM;
    #pragma unroll
    for (int kc = 0; kc < 4; ++kc) {
      const int k0 = kc * 32;
      a[kc] = *(const v16h*)(rp + k0 + 16*hi);
    }
  }

  // Row labels for the 8 C rows this lane sees (M = 8*hi + r).
  int rowlab[8];
  #pragma unroll
  for (int r = 0; r < 8; ++r)
    rowlab[r] = labels[(rowBase + 8*hi + r) & (BATCH - 1)];

  float denom[8], possum[8], cnt[8];
  #pragma unroll
  for (int r = 0; r < 8; ++r) { denom[r] = 0.f; possum[r] = 0.f; cnt[r] = 0.f; }

  for (int ct = 0; ct < NROWS; ct += CTILE) {
    __syncthreads();
    // Cooperative load of 64 columns (rows of F) into LDS: 1024 x 16B chunks.
    #pragma unroll
    for (int q = 0; q < 4; ++q) {
      const int chunk = t + q * 256;        // 0..1023
      const int cr = chunk >> 4;            // tile row 0..63
      const int co = (chunk & 15) * 8;      // half offset 0..120
      *(v8h*)(&Bsh[cr * DIM + co]) = *(const v8h*)(f16 + (size_t)(ct + cr) * DIM + co);
    }
    if (t < CTILE) labSh[t] = labels[(ct + t) & (BATCH - 1)];
    __syncthreads();

    // Prefetch next column tile into cache while we compute on this one.
    if (ct + CTILE < NROWS)
      __builtin_prefetch(f16 + (size_t)(ct + CTILE + (t >> 2)) * DIM + (t & 3) * 32, 0, 1);

    #pragma unroll
    for (int s = 0; s < 4; ++s) {
      v8f c = {0.f,0.f,0.f,0.f,0.f,0.f,0.f,0.f};
      #pragma unroll
      for (int kc = 0; kc < 4; ++kc) {
        // B layout: lane l (0-15) holds column N=l, contiguous K halves
        // [kc*32 + 16*hi .. +15] -> 32B LDS read.
        v16h b = *(const v16h*)(&Bsh[(s*16 + l2) * DIM + kc*32 + 16*hi]);
        c = __builtin_amdgcn_wmma_f32_16x16x32_f16(
                false, a[kc], false, b, (short)0, c, false, false);
      }
      const int j  = ct + s*16 + l2;
      const int jl = labSh[s*16 + l2];
      #pragma unroll
      for (int r = 0; r < 8; ++r) {
        const int i = rowBase + 8*hi + r;
        // Branchless: sel kills the diagonal, mat = sel & label-match.
        const float sel = (j != i) ? 1.0f : 0.0f;
        const float mat = (jl == rowlab[r]) ? sel : 0.0f;
        // exp(s/T - 1/T) computed as exp2(c*K2 - K2) -> one v_fma + v_exp_f32.
        const float e   = __builtin_amdgcn_exp2f(__builtin_fmaf(c[r], K2, -K2));
        const float sim = c[r] * INV_T;
        denom[r]  = __builtin_fmaf(sel, e,   denom[r]);
        possum[r] = __builtin_fmaf(mat, sim, possum[r]);
        cnt[r]   += mat;
      }
    }
  }

  // Reduce each row's stats across its 16 lanes (xor 8,4,2,1 stays in-group).
  #pragma unroll
  for (int r = 0; r < 8; ++r) {
    #pragma unroll
    for (int m = 8; m >= 1; m >>= 1) {
      denom[r]  += __shfl_xor(denom[r],  m, 32);
      possum[r] += __shfl_xor(possum[r], m, 32);
      cnt[r]    += __shfl_xor(cnt[r],    m, 32);
    }
  }
  if (l2 == 0) {
    #pragma unroll
    for (int r = 0; r < 8; ++r) {
      const float lse  = __logf(denom[r] + 1e-12f);
      const float cc   = fmaxf(cnt[r], 1.0f);
      // sum(mask*log_prob) = possum - cnt*(1/T + lse);  loss = -mean_pos
      const float mlpp = (possum[r] - cnt[r] * (INV_T + lse)) / cc;
      rowLoss[rowBase + 8*hi + r] = -mlpp;   // T/base_T == 1
    }
  }
}

// ---------------------------------------------------------------------------
// Kernel 3: mean of 8192 per-row losses -> scalar output.
// ---------------------------------------------------------------------------
__global__ __launch_bounds__(256) void supcon_reduce(
    const float* __restrict__ rowLoss, float* __restrict__ out) {
  __shared__ float sh[256];
  float s = 0.f;
  for (int k = threadIdx.x; k < NROWS; k += 256) s += rowLoss[k];
  sh[threadIdx.x] = s;
  __syncthreads();
  #pragma unroll
  for (int st = 128; st >= 1; st >>= 1) {
    if (threadIdx.x < st) sh[threadIdx.x] += sh[threadIdx.x + st];
    __syncthreads();
  }
  if (threadIdx.x == 0) out[0] = sh[0] * (1.0f / NROWS);
}

extern "C" void kernel_launch(void* const* d_in, const int* in_sizes, int n_in,
                              void* d_out, int out_size, void* d_ws, size_t ws_size,
                              hipStream_t stream) {
  const float* features = (const float*)d_in[0];   // [4096, 2, 128] f32
  const int*   labels   = (const int*)d_in[1];     // [4096] (int32 on device)
  float*       out      = (float*)d_out;           // scalar loss

  // Workspace: [0, 2MB) normalized f16 features; then 8192 f32 row losses.
  _Float16* f16 = (_Float16*)d_ws;
  float* rowLoss = (float*)((char*)d_ws + (size_t)NROWS * DIM * sizeof(_Float16));

  supcon_normalize<<<NROWS / 8, 256, 0, stream>>>(features, f16);
  supcon_main<<<NROWS / 128, 256, 0, stream>>>(f16, labels, rowLoss);
  supcon_reduce<<<1, 256, 0, stream>>>(rowLoss, out);
}
